// HierarchicalMultiHeadLatentAttention_38878043963935
// MI455X (gfx1250) — matrix operand order repaired and checked by reading, not compiled, not run on hardware
//
#include <hip/hip_runtime.h>

// ---------------------------------------------------------------------------
// Types for gfx1250 WMMA (wave32): 16x16x32 bf16 -> f32
// ---------------------------------------------------------------------------
typedef __attribute__((ext_vector_type(16))) __bf16 v16bf;
typedef __attribute__((ext_vector_type(8)))  float  v8f;

union Frag16 {
    v16bf v;
    uint4 q[2];
    unsigned short s[16];
};

__device__ __forceinline__ unsigned short f32_to_bf16_bits(float f) {
    unsigned u = __builtin_bit_cast(unsigned, f);
    unsigned r = u + 0x7FFFu + ((u >> 16) & 1u);   // round-to-nearest-even
    return (unsigned short)(r >> 16);
}

__device__ __forceinline__ v8f v8f_zero() {
    v8f z;
#pragma unroll
    for (int i = 0; i < 8; ++i) z[i] = 0.0f;
    return z;
}

// Async global->LDS copy of 32 bytes per lane (two B128 transfers).
// CDNA5 GLOBAL_LOAD_ASYNC_TO_LDS: INST_OFFSET is added on BOTH the LDS and
// the global side (CDNA5 ISA), so offset:16 covers the
// second 16B chunk on both addresses.  Tracked with ASYNCcnt.
__device__ __forceinline__ void async_copy_b128x2(const unsigned short* g,
                                                  unsigned short* lds_ptr) {
    unsigned lds_off = (unsigned)(uintptr_t)lds_ptr;       // addr[31:0] = LDS offset
    unsigned long long ga = (unsigned long long)(uintptr_t)g;
    asm volatile("global_load_async_to_lds_b128 %0, %1, off\n\t"
                 "global_load_async_to_lds_b128 %0, %1, off offset:16"
                 :: "v"(lds_off), "v"(ga) : "memory");
}

__device__ __forceinline__ void wait_async0() {
    asm volatile("s_wait_asynccnt 0x0" ::: "memory");
}

// ---------------------------------------------------------------------------
// Elementwise conversion kernels
// ---------------------------------------------------------------------------
__global__ __launch_bounds__(256) void conv_f32_bf16(const float* __restrict__ in,
                                                     unsigned short* __restrict__ out,
                                                     int n) {
    int i = blockIdx.x * 256 + threadIdx.x;
    if (i < n) out[i] = f32_to_bf16_bits(in[i]);
}

__global__ __launch_bounds__(256) void add_f32_bf16(const float* __restrict__ a,
                                                    const float* __restrict__ b,
                                                    unsigned short* __restrict__ out,
                                                    int n) {
    int i = blockIdx.x * 256 + threadIdx.x;
    if (i < n) out[i] = f32_to_bf16_bits(a[i] + b[i]);
}

// ---------------------------------------------------------------------------
// Tiled bf16 GEMM: C[M,N] = A[M,K] @ B[K,N]
// Block tile 128x128, BK=32, 256 threads = 8 waves; each wave owns a 16-row
// strip and 8 accumulators (16x16 tiles) -> 8 WMMAs per k-step per wave.
// A tile staged via async global->LDS (ASYNCcnt); B tile staged TRANSPOSED
// so both WMMA fragments are two 16B LDS loads per lane, matching the CDNA5
// 16-bit A / B VGPR layouts.  M,N % 128 == 0, K % 32 == 0 for all uses here.
// ---------------------------------------------------------------------------
template <int OUT_BF16>
__global__ __launch_bounds__(256) void gemm_bf16_kernel(
    const unsigned short* __restrict__ A, const unsigned short* __restrict__ B,
    void* __restrict__ C, int M, int N, int K) {
    __shared__ alignas(16) unsigned short As[128 * 32];   // [row][k]
    __shared__ alignas(16) unsigned short Bts[128 * 32];  // [n][k]  (transposed)

    const int tid  = threadIdx.x;
    const int wave = tid >> 5;
    const int lane = tid & 31;
    const int mtiles = M >> 7;
    const int bm = (blockIdx.x % mtiles) << 7;
    const int bn = (blockIdx.x / mtiles) << 7;

    v8f acc[8];
#pragma unroll
    for (int i = 0; i < 8; ++i) acc[i] = v8f_zero();

    const int a_row  = tid >> 1;          // 0..127
    const int a_half = (tid & 1) << 4;    // 0 or 16 elements
    const int b_k    = tid & 31;          // 0..31
    const int b_n0   = (tid >> 5) << 4;   // 0..112

    for (int k0 = 0; k0 < K; k0 += 32) {
        // ---- stage A tile (128x32 bf16): 32B per thread, async to LDS ----
        async_copy_b128x2(A + (size_t)(bm + a_row) * K + k0 + a_half,
                          As + a_row * 32 + a_half);

        // ---- stage B tile transposed (32x128 -> Bt[128][32]) ----
        {
            const uint4* src = (const uint4*)(B + (size_t)(k0 + b_k) * N + bn + b_n0);
            uint4 d0 = src[0], d1 = src[1];
            const unsigned short* ds0 = (const unsigned short*)&d0;
            const unsigned short* ds1 = (const unsigned short*)&d1;
#pragma unroll
            for (int j = 0; j < 8; ++j) {
                Bts[(b_n0 + j) * 32 + b_k]     = ds0[j];
                Bts[(b_n0 + 8 + j) * 32 + b_k] = ds1[j];
            }
        }
        if (k0 + 32 < K) {
            __builtin_prefetch(B + (size_t)(k0 + 32 + b_k) * N + bn + b_n0, 0, 0);
        }
        wait_async0();
        __syncthreads();

        // ---- A fragment: lane<16 -> row=lane, K {0..7,16..23}; lane>=16 -> K+8 ----
        Frag16 af;
        {
            const int row = wave * 16 + (lane & 15);
            const int kb  = (lane >> 4) << 3;  // element offset 0 or 8
            const uint4* p = (const uint4*)(As + row * 32 + kb);
            af.q[0] = p[0];   // K kb..kb+7
            af.q[1] = p[2];   // K kb+16..kb+23
        }
#pragma unroll
        for (int nt = 0; nt < 8; ++nt) {
            Frag16 bf_;
            const int n  = nt * 16 + (lane & 15);
            const int kb = (lane >> 4) << 3;   // 0 or 8
            const uint4* p = (const uint4*)(Bts + n * 32 + kb);
            bf_.q[0] = p[0];
            bf_.q[1] = p[2];
            acc[nt] = __builtin_amdgcn_wmma_f32_16x16x32_bf16(
                false, af.v, false, bf_.v, (short)0, acc[nt], false, false);
        }
        __syncthreads();
    }

    // ---- epilogue: C/D layout -> lane = col%16, elem = row (+8 for hi lanes) ----
#pragma unroll
    for (int nt = 0; nt < 8; ++nt) {
        const int col = bn + nt * 16 + (lane & 15);
#pragma unroll
        for (int r = 0; r < 8; ++r) {
            const int row = bm + wave * 16 + r + ((lane >> 4) << 3);
            const float v = acc[nt][r];
            if (OUT_BF16)
                ((unsigned short*)C)[(size_t)row * N + col] = f32_to_bf16_bits(v);
            else
                ((float*)C)[(size_t)row * N + col] = v;
        }
    }
}

// ---------------------------------------------------------------------------
// Flash attention with decay.  One block = (b, h, 128-row q strip); each wave
// owns 16 q rows and streams 32-key tiles.  QK^T fragments come straight from
// global (K rows already match the B^T fragment layout); P goes through
// per-wave LDS (D-layout -> A-layout); V is transposed into per-wave LDS.
// ---------------------------------------------------------------------------
__global__ __launch_bounds__(256) void attn_kernel(
    const unsigned short* __restrict__ Q, const unsigned short* __restrict__ Kr,
    const unsigned short* __restrict__ Vr, unsigned short* __restrict__ O) {
    constexpr int D = 1024, S = 2048;
    __shared__ alignas(16) unsigned short sP[8][16 * 32];   // per-wave P  (1KB)
    __shared__ alignas(16) unsigned short sVt[8][64 * 32];  // per-wave V^T (4KB)

    const int lane = threadIdx.x & 31;
    const int wave = threadIdx.x >> 5;
    const int blk  = blockIdx.x;
    const int qs   = blk & 15;          // S/128 = 16 strips
    const int h    = (blk >> 4) & 15;
    const int b    = blk >> 8;
    const int hoff = h * 64;
    const int qrow0 = qs * 128 + wave * 16;   // position within sequence
    const int grow0 = b * S + qrow0;          // global row

    // ---- preload q A-fragments for the two K=32 chunks of HD=64 ----
    Frag16 aq[2];
    {
        const int row  = grow0 + (lane & 15);
        const int koff = (lane >> 4) << 3;
        const unsigned short* base = Q + (size_t)row * D + hoff;
#pragma unroll
        for (int c = 0; c < 2; ++c) {
            const uint4* p = (const uint4*)(base + c * 32 + koff);
            aq[c].q[0] = p[0];
            aq[c].q[1] = p[2];
        }
    }

    v8f m_run, lsum, Oa[4];
#pragma unroll
    for (int e = 0; e < 8; ++e) { m_run[e] = -1e30f; lsum[e] = 0.0f; }
#pragma unroll
    for (int i = 0; i < 4; ++i) Oa[i] = v8f_zero();

    const float rowoff = (float)(qrow0 + ((lane >> 4) << 3));
    const float scale  = 0.03125f;  // 1/sqrt(D=1024)

    for (int kt = 0; kt < S / 32; ++kt) {
        const int k0 = kt * 32;

        // ---- stage V^T tile: lane owns key (k0+lane), writes its column ----
        {
            const int krow = b * S + k0 + lane;
            const uint4* src = (const uint4*)(Vr + (size_t)krow * D + hoff);
#pragma unroll
            for (int c = 0; c < 4; ++c) {
                uint4 d = src[c];
                const unsigned short* ds_ = (const unsigned short*)&d;
#pragma unroll
                for (int j = 0; j < 8; ++j) sVt[wave][(c * 8 + j) * 32 + lane] = ds_[j];
            }
        }

        // ---- scores: two 16x16 tiles over 32 keys, K-loop over HD (2x32) ----
        v8f s[2];
#pragma unroll
        for (int n = 0; n < 2; ++n) {
            v8f accv = v8f_zero();
            const int keyrow = b * S + k0 + n * 16 + (lane & 15);
            const unsigned short* kb_ =
                Kr + (size_t)keyrow * D + hoff + ((lane >> 4) << 3);
#pragma unroll
            for (int c = 0; c < 2; ++c) {
                Frag16 bf_;
                const uint4* p = (const uint4*)(kb_ + c * 32);
                bf_.q[0] = p[0];
                bf_.q[1] = p[2];
                accv = __builtin_amdgcn_wmma_f32_16x16x32_bf16(
                    false, aq[c].v, false, bf_.v, (short)0, accv, false, false);
            }
            s[n] = accv;
        }

        // ---- scale + decay ----
        const float tk0 = (float)(k0 + (lane & 15));
#pragma unroll
        for (int e = 0; e < 8; ++e) {
            const float tq = rowoff + (float)e;
            s[0][e] = s[0][e] * scale * __expf(-0.01f * fabsf(tq - tk0));
            s[1][e] = s[1][e] * scale * __expf(-0.01f * fabsf(tq - (tk0 + 16.0f)));
        }

        // ---- online softmax (row reductions across 16-lane groups) ----
        v8f mx;
#pragma unroll
        for (int e = 0; e < 8; ++e) mx[e] = fmaxf(s[0][e], s[1][e]);
#pragma unroll
        for (int off = 1; off < 16; off <<= 1) {
#pragma unroll
            for (int e = 0; e < 8; ++e) mx[e] = fmaxf(mx[e], __shfl_xor(mx[e], off));
        }

        v8f rs;
#pragma unroll
        for (int e = 0; e < 8; ++e) {
            const float mnew = fmaxf(m_run[e], mx[e]);
            const float corr = __expf(m_run[e] - mnew);
            const float p0 = __expf(s[0][e] - mnew);
            const float p1 = __expf(s[1][e] - mnew);
            m_run[e] = mnew;
            lsum[e] *= corr;
            rs[e] = p0 + p1;
#pragma unroll
            for (int cc = 0; cc < 4; ++cc) Oa[cc][e] *= corr;
            const int prow = e + ((lane >> 4) << 3);
            sP[wave][prow * 32 + (lane & 15)]      = f32_to_bf16_bits(p0);
            sP[wave][prow * 32 + 16 + (lane & 15)] = f32_to_bf16_bits(p1);
        }
#pragma unroll
        for (int off = 1; off < 16; off <<= 1) {
#pragma unroll
            for (int e = 0; e < 8; ++e) rs[e] += __shfl_xor(rs[e], off);
        }
#pragma unroll
        for (int e = 0; e < 8; ++e) lsum[e] += rs[e];

        // per-wave private LDS; DS ops are in-order per wave — just pin ordering
        asm volatile("s_wait_dscnt 0x0" ::: "memory");

        // ---- O += P @ V  (A = P 16x32, B = V^T fragments) ----
        Frag16 ap;
        {
            const int prow = lane & 15;
            const int kb   = (lane >> 4) << 3;
            const uint4* p = (const uint4*)(sP[wave] + prow * 32 + kb);
            ap.q[0] = p[0];
            ap.q[1] = p[2];
        }
#pragma unroll
        for (int cc = 0; cc < 4; ++cc) {
            Frag16 bv;
            const int n  = cc * 16 + (lane & 15);
            const int kb = (lane >> 4) << 3;
            const uint4* p = (const uint4*)(sVt[wave] + n * 32 + kb);
            bv.q[0] = p[0];
            bv.q[1] = p[2];
            Oa[cc] = __builtin_amdgcn_wmma_f32_16x16x32_bf16(
                false, ap.v, false, bv.v, (short)0, Oa[cc], false, false);
        }
        asm volatile("" ::: "memory");
    }

    // ---- normalize + store bf16 ----
#pragma unroll
    for (int e = 0; e < 8; ++e) {
        const float inv = 1.0f / lsum[e];
        const int row = grow0 + e + ((lane >> 4) << 3);
#pragma unroll
        for (int cc = 0; cc < 4; ++cc) {
            O[(size_t)row * D + hoff + cc * 16 + (lane & 15)] =
                f32_to_bf16_bits(Oa[cc][e] * inv);
        }
    }
}

// ---------------------------------------------------------------------------
// Host-side orchestration
// ---------------------------------------------------------------------------
extern "C" void kernel_launch(void* const* d_in, const int* in_sizes, int n_in,
                              void* d_out, int out_size, void* d_ws, size_t ws_size,
                              hipStream_t stream) {
    (void)in_sizes; (void)n_in; (void)out_size; (void)ws_size;
    constexpr int B = 4, S = 2048, D = 1024, DL = 256;
    constexpr int M = B * S;  // 8192

    const float* x      = (const float*)d_in[0];
    const float* W_q    = (const float*)d_in[1];
    const float* W_k    = (const float*)d_in[2];
    const float* W_v    = (const float*)d_in[3];
    const float* W_o    = (const float*)d_in[4];
    const float* W_down = (const float*)d_in[5];
    const float* W_upk  = (const float*)d_in[6];
    const float* W_upv  = (const float*)d_in[7];

    char* ws = (char*)d_ws;
    size_t off = 0;
    auto alloc = [&](size_t bytes) { char* p = ws + off; off += (bytes + 255) & ~(size_t)255; return p; };

    unsigned short* xb     = (unsigned short*)alloc((size_t)M * D * 2);
    unsigned short* Wq_b   = (unsigned short*)alloc((size_t)D * D * 2);
    unsigned short* Wo_b   = (unsigned short*)alloc((size_t)D * D * 2);
    unsigned short* Wkv_b  = (unsigned short*)alloc((size_t)D * D * 2);
    unsigned short* Wdn_b  = (unsigned short*)alloc((size_t)D * DL * 2);
    unsigned short* Wupk_b = (unsigned short*)alloc((size_t)DL * D * 2);
    unsigned short* Wupv_b = (unsigned short*)alloc((size_t)DL * D * 2);
    unsigned short* Wcomb  = (unsigned short*)alloc((size_t)D * DL * 2);
    unsigned short* qb     = (unsigned short*)alloc((size_t)M * D * 2);
    unsigned short* latb   = (unsigned short*)alloc((size_t)M * DL * 2);
    unsigned short* krecb  = (unsigned short*)alloc((size_t)M * D * 2);
    unsigned short* vrecb  = (unsigned short*)alloc((size_t)M * D * 2);
    unsigned short* attnb  = (unsigned short*)alloc((size_t)M * D * 2);

    auto cgrid = [](int n) { return dim3((n + 255) / 256); };

    // ---- precision conversion + weight fusion ----
    conv_f32_bf16<<<cgrid(M * D), 256, 0, stream>>>(x, xb, M * D);
    conv_f32_bf16<<<cgrid(D * D), 256, 0, stream>>>(W_q, Wq_b, D * D);
    conv_f32_bf16<<<cgrid(D * D), 256, 0, stream>>>(W_o, Wo_b, D * D);
    conv_f32_bf16<<<cgrid(D * DL), 256, 0, stream>>>(W_down, Wdn_b, D * DL);
    conv_f32_bf16<<<cgrid(DL * D), 256, 0, stream>>>(W_upk, Wupk_b, DL * D);
    conv_f32_bf16<<<cgrid(DL * D), 256, 0, stream>>>(W_upv, Wupv_b, DL * D);
    add_f32_bf16<<<cgrid(D * D), 256, 0, stream>>>(W_k, W_v, Wkv_b, D * D);

    // W_comb = (W_k + W_v) @ W_down   [1024,256]
    gemm_bf16_kernel<1><<<dim3((D / 128) * (DL / 128)), 256, 0, stream>>>(
        Wkv_b, Wdn_b, Wcomb, D, DL, D);

    // q = x @ W_q                     [8192,1024]
    gemm_bf16_kernel<1><<<dim3((M / 128) * (D / 128)), 256, 0, stream>>>(
        xb, Wq_b, qb, M, D, D);

    // latent = x @ W_comb             [8192,256]
    gemm_bf16_kernel<1><<<dim3((M / 128) * (DL / 128)), 256, 0, stream>>>(
        xb, Wcomb, latb, M, DL, D);

    // k_rec = latent @ W_up_k, v_rec = latent @ W_up_v   [8192,1024]
    gemm_bf16_kernel<1><<<dim3((M / 128) * (D / 128)), 256, 0, stream>>>(
        latb, Wupk_b, krecb, M, D, DL);
    gemm_bf16_kernel<1><<<dim3((M / 128) * (D / 128)), 256, 0, stream>>>(
        latb, Wupv_b, vrecb, M, D, DL);

    // attention: B*H*(S/128) = 4*16*16 = 1024 blocks
    attn_kernel<<<dim3(1024), 256, 0, stream>>>(qb, krecb, vrecb, attnb);

    // out = attn_out @ W_o  (f32 to d_out)
    gemm_bf16_kernel<0><<<dim3((M / 128) * (D / 128)), 256, 0, stream>>>(
        attnb, Wo_b, d_out, M, D, D);
}
